// DeepseekV2MoE_65515431133681
// MI455X (gfx1250) — matrix operand order repaired and checked by reading, not compiled, not run on hardware
//
#include <hip/hip_runtime.h>

// ---------------- problem constants (match reference) ----------------
#define T_TOK 4096
#define HDIM  2048
#define NEXP  64
#define TOPK  6
#define IDIM  512
#define CAP   1024
#define SHI   1024   // shared experts intermediate (2*I)
#define NGRP  8
#define TGRP  3
#define GSZ   (NEXP / NGRP)
#define RSF   2.5f

// ---------------- WMMA tiling ----------------
#define BM 128
#define BN 64
#define BK 32
#define NTHREADS 256

typedef __attribute__((ext_vector_type(16))) __bf16 v16bf;
typedef __attribute__((ext_vector_type(8)))  float  v8f;

union Frag16 { v16bf v; uint4 q[2]; };
union Pack8  { __bf16 h[8]; uint4 q; };

// -------- CDNA5 async memory->LDS copy (ASYNCcnt path), with fallback --------
#if defined(__has_builtin)
#  if __has_builtin(__builtin_amdgcn_global_load_async_to_lds_b128)
#    define USE_ASYNC_LDS 1
#  endif
#endif

#ifdef USE_ASYNC_LDS
// builtin expects: (int __vector(4) __device__*, int __vector(4) __shared__*, imm, imm)
typedef int async_v4i __attribute__((vector_size(16)));
#endif

__device__ __forceinline__ void copy16_to_lds(const __bf16* g, __bf16* l) {
#ifdef USE_ASYNC_LDS
  __builtin_amdgcn_global_load_async_to_lds_b128(
      (__attribute__((address_space(1))) async_v4i*)(void*)g,
      (__attribute__((address_space(3))) async_v4i*)(void*)l, 0, 0);
#else
  *(uint4*)l = *(const uint4*)g;
#endif
}
__device__ __forceinline__ void wait_async_copies() {
#ifdef USE_ASYNC_LDS
  asm volatile("s_wait_asynccnt 0x0" ::: "memory");
#endif
}

// ---------------- fp32 -> bf16 convert ----------------
__global__ void cvt_bf16_kernel(const float* __restrict__ x, __bf16* __restrict__ y, long n) {
  long i = (long)blockIdx.x * blockDim.x + threadIdx.x;
  long stride = (long)gridDim.x * blockDim.x;
  for (; i < n; i += stride) y[i] = (__bf16)x[i];
}

// ---------------- router: softmax + grouped top-k ----------------
__global__ __launch_bounds__(64)
void router_kernel(const float* __restrict__ x, const float* __restrict__ gw,
                   float* __restrict__ topk_w, int* __restrict__ topk_ids) {
  __shared__ float xs[HDIM];
  __shared__ float sc[NEXP];
  const int t = blockIdx.x;
  for (int i = threadIdx.x; i < HDIM; i += 64) xs[i] = x[(size_t)t * HDIM + i];
  __syncthreads();
  {
    const float* wr = gw + (size_t)threadIdx.x * HDIM;
    float acc = 0.f;
    #pragma unroll 4
    for (int i = 0; i < HDIM; ++i) acc += xs[i] * wr[i];
    sc[threadIdx.x] = acc;
  }
  __syncthreads();
  if (threadIdx.x == 0) {
    float m = sc[0];
    for (int e = 1; e < NEXP; ++e) m = fmaxf(m, sc[e]);
    float s = 0.f;
    for (int e = 0; e < NEXP; ++e) { sc[e] = __expf(sc[e] - m); s += sc[e]; }
    float inv = 1.f / s;
    for (int e = 0; e < NEXP; ++e) sc[e] *= inv;
    float gmax[NGRP];
    for (int g = 0; g < NGRP; ++g) {
      float mg = sc[g * GSZ];
      for (int j = 1; j < GSZ; ++j) mg = fmaxf(mg, sc[g * GSZ + j]);
      gmax[g] = mg;
    }
    bool gsel[NGRP];
    for (int g = 0; g < NGRP; ++g) gsel[g] = false;
    for (int it = 0; it < TGRP; ++it) {
      int best = 0; float bv = -1.f;
      for (int g = 0; g < NGRP; ++g)
        if (!gsel[g] && gmax[g] > bv) { bv = gmax[g]; best = g; }
      gsel[best] = true;
    }
    for (int e = 0; e < NEXP; ++e) if (!gsel[e / GSZ]) sc[e] = -1.f;
    float wsum = 0.f;
    int ids[TOPK]; float ws[TOPK];
    for (int k = 0; k < TOPK; ++k) {
      int best = 0; float bv = -2.f;
      for (int e = 0; e < NEXP; ++e) if (sc[e] > bv) { bv = sc[e]; best = e; }
      ids[k] = best; ws[k] = fmaxf(bv, 0.f); wsum += ws[k]; sc[best] = -3.f;
    }
    float invw = 1.f / wsum;
    for (int k = 0; k < TOPK; ++k) {
      topk_w  [(size_t)t * TOPK + k] = ws[k] * invw;
      topk_ids[(size_t)t * TOPK + k] = ids[k];
    }
  }
}

// ---------------- dispatch ----------------
__global__ void dispatch_init_kernel(int* __restrict__ cnt, int* __restrict__ slot_token,
                                     __bf16* __restrict__ zpad) {
  int i = blockIdx.x * blockDim.x + threadIdx.x;
  if (i < NEXP) cnt[i] = 0;
  if (blockIdx.x == 0 && threadIdx.x < 32) zpad[threadIdx.x] = (__bf16)0.f;
  int stride = gridDim.x * blockDim.x;
  for (int s = i; s < NEXP * CAP; s += stride) slot_token[s] = -1;
}

__global__ void assign_kernel(const int* __restrict__ topk_ids, const float* __restrict__ topk_w,
                              int* __restrict__ cnt, int* __restrict__ slot_token,
                              float* __restrict__ slot_wt) {
  int j = blockIdx.x * blockDim.x + threadIdx.x;
  if (j >= T_TOK * TOPK) return;
  int e = topk_ids[j];
  int pos = atomicAdd(&cnt[e], 1);
  if (pos < CAP) {
    slot_token[e * CAP + pos] = j / TOPK;
    slot_wt  [e * CAP + pos] = topk_w[j] * RSF;
  }
}

// ---------------- fused gate_up GEMM + SiLU*mul (WMMA bf16, double-buffered) ----
// out_act[row, 0:Iout] = silu(A @ Wg^T) * (A @ Wu^T),  W = [2*Iout, Kdim] fp32 rows
// slot_token != nullptr => routed-expert mode (A-row gather, capacity early-exit)
__global__ __launch_bounds__(NTHREADS)
void gateup_silu_kernel(const __bf16* __restrict__ Abf, const float* __restrict__ Wbase,
                        __bf16* __restrict__ actout,
                        const int* __restrict__ slot_token, const int* __restrict__ cnt,
                        const __bf16* __restrict__ zpad,
                        int Kdim, int Iout) {
  __shared__ __bf16 As [2][BM][BK];
  __shared__ __bf16 Bsg[2][BN][BK];
  __shared__ __bf16 Bsu[2][BN][BK];
  const int e     = blockIdx.z;
  const int mBase = blockIdx.y * BM;
  const int nb    = blockIdx.x * BN;
  if (cnt) {
    int c = cnt[e]; if (c > CAP) c = CAP;
    if (mBase >= c) return;
  }
  const float* W = Wbase + (size_t)e * (size_t)(2 * Iout) * (size_t)Kdim;
  const int tid = threadIdx.x;
  const int lane = tid & 31, wv = tid >> 5;
  const int mw = wv >> 2, nw = wv & 3;          // 2 x 4 waves
  const int r = lane & 15, hi = lane >> 4;

  v8f cg[4], cu[4];
  v8f zero = {0.f,0.f,0.f,0.f,0.f,0.f,0.f,0.f};
  #pragma unroll
  for (int mi = 0; mi < 4; ++mi) { cg[mi] = zero; cu[mi] = zero; }

  const int arow = tid >> 1, ahalf = tid & 1;   // A tile: 128 rows x 2 halves of 16 bf16
  const int brow = tid >> 2, bq = tid & 3;      // B tile: 64 rows x 4 chunks of 8 floats

  // A source: direct row, gathered row, or zero pad
  const __bf16* aSrc; int aStep;
  if (slot_token) {
    int tok = slot_token[e * CAP + mBase + arow];
    if (tok >= 0) { aSrc = Abf + (size_t)tok * Kdim + ahalf * 16; aStep = BK; }
    else          { aSrc = zpad;                                  aStep = 0;  }
  } else {
    aSrc = Abf + (size_t)(mBase + arow) * Kdim + ahalf * 16; aStep = BK;
  }
  const float* bgp = W + (size_t)(nb + brow) * Kdim + bq * 8;
  const float* bup = W + (size_t)(Iout + nb + brow) * Kdim + bq * 8;

  // ---- prologue: stage tile 0 ----
  copy16_to_lds(aSrc + 0, &As[0][arow][ahalf * 16 + 0]);
  copy16_to_lds(aSrc + 8, &As[0][arow][ahalf * 16 + 8]);
  aSrc += aStep;
  float4 g0 = *(const float4*)bgp, g1 = *(const float4*)(bgp + 4);
  float4 u0 = *(const float4*)bup, u1 = *(const float4*)(bup + 4);
  bgp += BK; bup += BK;

  const int ntiles = Kdim / BK;
  for (int i = 0; i < ntiles; ++i) {
    const int buf = i & 1;
    wait_async_copies();                 // my A copies for tile i are in LDS
    {                                    // commit staged B regs (fp32 -> bf16)
      Pack8 pg, pu;
      pg.h[0]=(__bf16)g0.x; pg.h[1]=(__bf16)g0.y; pg.h[2]=(__bf16)g0.z; pg.h[3]=(__bf16)g0.w;
      pg.h[4]=(__bf16)g1.x; pg.h[5]=(__bf16)g1.y; pg.h[6]=(__bf16)g1.z; pg.h[7]=(__bf16)g1.w;
      pu.h[0]=(__bf16)u0.x; pu.h[1]=(__bf16)u0.y; pu.h[2]=(__bf16)u0.z; pu.h[3]=(__bf16)u0.w;
      pu.h[4]=(__bf16)u1.x; pu.h[5]=(__bf16)u1.y; pu.h[6]=(__bf16)u1.z; pu.h[7]=(__bf16)u1.w;
      *(uint4*)&Bsg[buf][brow][bq * 8] = pg.q;
      *(uint4*)&Bsu[buf][brow][bq * 8] = pu.q;
    }
    __syncthreads();                     // tile i fully staged; bank buf^1 free
    if (i + 1 < ntiles) {                // stage tile i+1 into the other bank
      copy16_to_lds(aSrc + 0, &As[buf ^ 1][arow][ahalf * 16 + 0]);
      copy16_to_lds(aSrc + 8, &As[buf ^ 1][arow][ahalf * 16 + 8]);
      aSrc += aStep;
      g0 = *(const float4*)bgp; g1 = *(const float4*)(bgp + 4);
      u0 = *(const float4*)bup; u1 = *(const float4*)(bup + 4);
      bgp += BK; bup += BK;
      __builtin_prefetch(bgp, 0, 1);
      __builtin_prefetch(bup, 0, 1);
    }
    // ---- compute tile i ----
    Frag16 fbg, fbu;
    fbg.q[0] = *(const uint4*)&Bsg[buf][nw * 16 + r][hi * 16 + 0];
    fbg.q[1] = *(const uint4*)&Bsg[buf][nw * 16 + r][hi * 16 + 8];
    fbu.q[0] = *(const uint4*)&Bsu[buf][nw * 16 + r][hi * 16 + 0];
    fbu.q[1] = *(const uint4*)&Bsu[buf][nw * 16 + r][hi * 16 + 8];
    #pragma unroll
    for (int mi = 0; mi < 4; ++mi) {
      Frag16 fa;
      fa.q[0] = *(const uint4*)&As[buf][mw * 64 + mi * 16 + r][hi * 16 + 0];
      fa.q[1] = *(const uint4*)&As[buf][mw * 64 + mi * 16 + r][hi * 16 + 8];
      cg[mi] = __builtin_amdgcn_wmma_f32_16x16x32_bf16(false, fa.v, false, fbg.v,
                                                       (short)0, cg[mi], false, false);
      cu[mi] = __builtin_amdgcn_wmma_f32_16x16x32_bf16(false, fa.v, false, fbu.v,
                                                       (short)0, cu[mi], false, false);
    }
  }

  // ---- epilogue: act = silu(g)*u -> bf16 ----
  const size_t aOutBase = (slot_token ? (size_t)e * CAP : (size_t)0) + (size_t)mBase;
  const int col = nb + nw * 16 + r;
  #pragma unroll
  for (int mi = 0; mi < 4; ++mi) {
    const int rowInTile = mw * 64 + mi * 16 + 8 * hi;
    #pragma unroll
    for (int v = 0; v < 8; ++v) {
      float g = cg[mi][v], u = cu[mi][v];
      float act = g / (1.f + __expf(-g)) * u;
      actout[(aOutBase + rowInTile + v) * (size_t)Iout + col] = (__bf16)act;
    }
  }
}

// ---------------- down-proj GEMM (WMMA bf16, double-buffered) + combine --------
// shared mode (slot_token==nullptr): out[row, :] = A @ W^T           (plain store)
// expert mode: atomicAdd(out[tok, :], (A @ W^T)[slot, :] * slot_wt)  (scaled combine)
__global__ __launch_bounds__(NTHREADS)
void down_combine_kernel(const __bf16* __restrict__ Aact, const float* __restrict__ Wbase,
                         float* __restrict__ out,
                         const int* __restrict__ slot_token, const float* __restrict__ slot_wt,
                         const int* __restrict__ cnt,
                         int Kdim, int Nout) {
  __shared__ __bf16 As[2][BM][BK];
  __shared__ __bf16 Bs[2][BN][BK];
  const int e     = blockIdx.z;
  const int mBase = blockIdx.y * BM;
  const int nb    = blockIdx.x * BN;
  if (cnt) {
    int c = cnt[e]; if (c > CAP) c = CAP;
    if (mBase >= c) return;
  }
  const float* W = Wbase + (size_t)e * (size_t)Nout * (size_t)Kdim;
  const int tid = threadIdx.x;
  const int lane = tid & 31, wv = tid >> 5;
  const int mw = wv >> 2, nw = wv & 3;
  const int r = lane & 15, hi = lane >> 4;

  v8f c[4];
  v8f zero = {0.f,0.f,0.f,0.f,0.f,0.f,0.f,0.f};
  #pragma unroll
  for (int mi = 0; mi < 4; ++mi) c[mi] = zero;

  const int arow = tid >> 1, ahalf = tid & 1;
  const int brow = tid >> 2, bq = tid & 3;
  const size_t aRowBase = (slot_token ? (size_t)e * CAP : (size_t)0) + (size_t)mBase;

  const __bf16* aSrc = Aact + (aRowBase + arow) * (size_t)Kdim + ahalf * 16;
  const float*  bp   = W + (size_t)(nb + brow) * Kdim + bq * 8;

  // ---- prologue: stage tile 0 ----
  copy16_to_lds(aSrc + 0, &As[0][arow][ahalf * 16 + 0]);
  copy16_to_lds(aSrc + 8, &As[0][arow][ahalf * 16 + 8]);
  aSrc += BK;
  float4 b0 = *(const float4*)bp, b1 = *(const float4*)(bp + 4);
  bp += BK;

  const int ntiles = Kdim / BK;
  for (int i = 0; i < ntiles; ++i) {
    const int buf = i & 1;
    wait_async_copies();
    {
      Pack8 p;
      p.h[0]=(__bf16)b0.x; p.h[1]=(__bf16)b0.y; p.h[2]=(__bf16)b0.z; p.h[3]=(__bf16)b0.w;
      p.h[4]=(__bf16)b1.x; p.h[5]=(__bf16)b1.y; p.h[6]=(__bf16)b1.z; p.h[7]=(__bf16)b1.w;
      *(uint4*)&Bs[buf][brow][bq * 8] = p.q;
    }
    __syncthreads();
    if (i + 1 < ntiles) {
      copy16_to_lds(aSrc + 0, &As[buf ^ 1][arow][ahalf * 16 + 0]);
      copy16_to_lds(aSrc + 8, &As[buf ^ 1][arow][ahalf * 16 + 8]);
      aSrc += BK;
      b0 = *(const float4*)bp; b1 = *(const float4*)(bp + 4);
      bp += BK;
      __builtin_prefetch(bp, 0, 1);
    }
    Frag16 fb;
    fb.q[0] = *(const uint4*)&Bs[buf][nw * 16 + r][hi * 16 + 0];
    fb.q[1] = *(const uint4*)&Bs[buf][nw * 16 + r][hi * 16 + 8];
    #pragma unroll
    for (int mi = 0; mi < 4; ++mi) {
      Frag16 fa;
      fa.q[0] = *(const uint4*)&As[buf][mw * 64 + mi * 16 + r][hi * 16 + 0];
      fa.q[1] = *(const uint4*)&As[buf][mw * 64 + mi * 16 + r][hi * 16 + 8];
      c[mi] = __builtin_amdgcn_wmma_f32_16x16x32_bf16(false, fa.v, false, fb.v,
                                                      (short)0, c[mi], false, false);
    }
  }

  const int col = nb + nw * 16 + r;
  #pragma unroll
  for (int mi = 0; mi < 4; ++mi) {
    const int rowInTile = mw * 64 + mi * 16 + 8 * hi;
    #pragma unroll
    for (int v = 0; v < 8; ++v) {
      float val = c[mi][v];
      if (slot_token) {
        int s   = mBase + rowInTile + v;
        int tok = slot_token[e * CAP + s];
        if (tok >= 0)
          atomicAdd(out + (size_t)tok * Nout + col, val * slot_wt[e * CAP + s]);
      } else {
        out[(size_t)(mBase + rowInTile + v) * Nout + col] = val;
      }
    }
  }
}

// ---------------- host launch ----------------
extern "C" void kernel_launch(void* const* d_in, const int* in_sizes, int n_in,
                              void* d_out, int out_size, void* d_ws, size_t ws_size,
                              hipStream_t stream) {
  (void)in_sizes; (void)n_in; (void)out_size; (void)ws_size;
  const float* x    = (const float*)d_in[0];   // [T, H]
  const float* gw   = (const float*)d_in[1];   // [E, H]
  const float* w13  = (const float*)d_in[2];   // [E, 2I, H]
  const float* w2   = (const float*)d_in[3];   // [E, H, I]
  const float* swgu = (const float*)d_in[4];   // [2*SHI, H]
  const float* swd  = (const float*)d_in[5];   // [H, SHI]
  float* out = (float*)d_out;                  // [T, H]

  char* ws = (char*)d_ws;
  size_t off = 0;
  auto carve = [&](size_t bytes) -> void* {
    void* p = ws + off;
    off += (bytes + 255) & ~(size_t)255;
    return p;
  };
  __bf16* xbf   = (__bf16*)carve((size_t)T_TOK * HDIM * 2);       // 16.8 MB
  __bf16* shact = (__bf16*)carve((size_t)T_TOK * SHI  * 2);       //  8.4 MB
  __bf16* act   = (__bf16*)carve((size_t)NEXP * CAP * IDIM * 2);  // 67.1 MB
  float*  tw    = (float*) carve((size_t)T_TOK * TOPK * 4);
  int*    tids  = (int*)   carve((size_t)T_TOK * TOPK * 4);
  int*    cnt   = (int*)   carve((size_t)NEXP * 4);
  int*    stok  = (int*)   carve((size_t)NEXP * CAP * 4);
  float*  swt   = (float*) carve((size_t)NEXP * CAP * 4);
  __bf16* zpad  = (__bf16*)carve(64);                             // zero rows for gather miss

  // 1) hidden states -> bf16 (shared by shared-expert and routed-expert GEMMs)
  cvt_bf16_kernel<<<4096, 256, 0, stream>>>(x, xbf, (long)T_TOK * HDIM);
  // 2) router
  router_kernel<<<T_TOK, 64, 0, stream>>>(x, gw, tw, tids);
  // 3) dispatch
  dispatch_init_kernel<<<256, 256, 0, stream>>>(cnt, stok, zpad);
  assign_kernel<<<(T_TOK * TOPK + 255) / 256, 256, 0, stream>>>(tids, tw, cnt, stok, swt);
  // 4) shared experts: gate_up + silu*mul
  gateup_silu_kernel<<<dim3(SHI / BN, T_TOK / BM, 1), NTHREADS, 0, stream>>>(
      xbf, swgu, shact, nullptr, nullptr, zpad, HDIM, SHI);
  // 5) routed experts: gate_up + silu*mul (gathered rows, capacity early-exit)
  gateup_silu_kernel<<<dim3(IDIM / BN, CAP / BM, NEXP), NTHREADS, 0, stream>>>(
      xbf, w13, act, stok, cnt, zpad, HDIM, IDIM);
  // 6) shared experts down-proj: initializes d_out
  down_combine_kernel<<<dim3(HDIM / BN, T_TOK / BM, 1), NTHREADS, 0, stream>>>(
      shact, swd, out, nullptr, nullptr, nullptr, SHI, HDIM);
  // 7) routed experts down-proj: scaled atomic combine into d_out
  down_combine_kernel<<<dim3(HDIM / BN, CAP / BM, NEXP), NTHREADS, 0, stream>>>(
      act, w2, out, stok, swt, cnt, IDIM, HDIM);
}
